// BasicBlock_55654186221980
// MI455X (gfx1250) — 8-bit matrix operand order repaired and checked by reading, compiled, not run on hardware
//
#include <hip/hip_runtime.h>

typedef __attribute__((ext_vector_type(16))) int   v16i;
typedef __attribute__((ext_vector_type(8)))  float v8f;

#define E_DIM 512
#define B_SZ  16
#define CIN   256
#define COUT  128
#define HDIM  64
#define WDIM  64
#define HP    66
#define WP    66

#define OFF_SILU   0u
#define OFF_BIAS1  32768u
#define OFF_BIAS2  49152u
#define OFF_BIAS3  57344u
#define OFF_ALPHA  65536u
#define OFF_BETA   66048u
#define OFF_WPACK  66560u
#define OFF_ACTP   361472u
#define ACTP_BYTES (16u*66u*66u*256u)

__global__ void k_silu(const float* __restrict__ emb, float* __restrict__ s) {
    int i = blockIdx.x * 256 + threadIdx.x;
    if (i < B_SZ * E_DIM) {
        float e = emb[i];
        s[i] = e / (1.f + __expf(-e));
    }
}

__global__ void k_bias(const float* __restrict__ s,
                       const float* __restrict__ m1w, const float* __restrict__ m1b,
                       const float* __restrict__ m2w, const float* __restrict__ m2b,
                       const float* __restrict__ m3w, const float* __restrict__ m3b,
                       float* __restrict__ bias1, float* __restrict__ bias2,
                       float* __restrict__ bias3) {
    int t = blockIdx.x * 256 + threadIdx.x;
    if (t < 4096) {
        int b = t >> 8, c = t & 255;
        const float* sv = s + b * E_DIM;
        const float* wr = m1w + c * E_DIM;
        float acc = m1b[c];
        for (int e = 0; e < E_DIM; ++e) acc += sv[e] * wr[e];
        bias1[t] = acc;
    } else if (t < 6144) {
        int idx = t - 4096;
        int b = idx >> 7, c = idx & 127;
        const float* sv = s + b * E_DIM;
        const float* wr = m2w + c * E_DIM;
        float acc = m2b[c];
        for (int e = 0; e < E_DIM; ++e) acc += sv[e] * wr[e];
        bias2[idx] = acc;
    } else if (t < 8192) {
        int idx = t - 6144;
        int b = idx >> 7, c = idx & 127;
        const float* sv = s + b * E_DIM;
        const float* wr = m3w + c * E_DIM;
        float acc = m3b[c];
        for (int e = 0; e < E_DIM; ++e) acc += sv[e] * wr[e];
        bias3[idx] = acc;
    }
}

__global__ void k_scale(const float* __restrict__ conv_w, const float* __restrict__ conv_b,
                        const float* __restrict__ gam, const float* __restrict__ bet,
                        const float* __restrict__ mean, const float* __restrict__ var,
                        float* __restrict__ alpha, float* __restrict__ beta) {
    __shared__ float red[256];
    int co = blockIdx.x;
    float acc = 0.f;
    for (int i = threadIdx.x; i < CIN * 9; i += 256)
        acc += fabsf(conv_w[co * CIN * 9 + i]);
    red[threadIdx.x] = acc;
    __syncthreads();
    for (int s = 128; s > 0; s >>= 1) {
        if (threadIdx.x < s) red[threadIdx.x] += red[threadIdx.x + s];
        __syncthreads();
    }
    if (threadIdx.x == 0) {
        float scale = red[0] / (float)(CIN * 9);
        float inv = gam[co] * rsqrtf(var[co] + 1e-5f);
        alpha[co] = scale * inv;
        beta[co] = (conv_b[co] - mean[co]) * inv + bet[co];
    }
}

__global__ void k_wpack(const float* __restrict__ conv_w, unsigned char* __restrict__ wpack) {
    int f = blockIdx.x * 256 + threadIdx.x;
    if (f >= 294912) return;
    int byte = f & 63;
    int lane = (f >> 6) & 31;
    int frag = f >> 11;
    int cotile = frag & 7;
    int q = (frag >> 3) & 1;
    int t = frag >> 4;
    int v = byte >> 2, j = byte & 3;
    int K = 32 * (v >> 2) + 16 * (lane >> 4) + 4 * (v & 3) + j;
    int co = (cotile << 4) + (lane & 15);
    int ci = (q << 7) + K;
    float wv = conv_w[(co * CIN + ci) * 9 + t];
    wpack[f] = wv > 0.f ? 0x38 : (wv < 0.f ? 0xB8 : 0x00);
}

__global__ void k_fill(int4* __restrict__ p, int n16) {
    int i = blockIdx.x * 256 + threadIdx.x;
    if (i < n16) p[i] = make_int4(0, 0, 0, 0);
}

__global__ void k_binarize(const float* __restrict__ x, const float* __restrict__ bias1,
                           unsigned char* __restrict__ actp) {
    int pix = blockIdx.x;
    int ci = threadIdx.x;
    int b = pix >> 12;
    int h = (pix >> 6) & 63;
    int w = pix & 63;
    float v = x[(((b << 8) + ci) << 12) + (h << 6) + w] + bias1[(b << 8) + ci];
    unsigned char s = v > 0.f ? 0x38 : (v < 0.f ? 0xB8 : 0x00);
    actp[(((b * HP + h + 1) * WP + (w + 1)) << 8) + ci] = s;
}

__global__ __launch_bounds__(256) void k_gemm(
    const unsigned char* __restrict__ actp, const unsigned char* __restrict__ wpack,
    const float* __restrict__ alpha, const float* __restrict__ beta,
    const float* __restrict__ bias2, const float* __restrict__ bias3,
    const float* __restrict__ prelu_a, const float* __restrict__ x,
    float* __restrict__ out) {
    const int lane = threadIdx.x & 31;
    const int cotile = threadIdx.x >> 5;
    const int ntile = blockIdx.x;
    const int b = ntile >> 8;
    const int rem = ntile & 255;
    const int h = rem >> 2;
    const int w0 = (rem & 3) << 4;
    const int n = lane & 15;
    const int hi = lane >> 4;
    const int w = w0 + n;

    v8f c = {0.f, 0.f, 0.f, 0.f, 0.f, 0.f, 0.f, 0.f};

    const unsigned char* wbase = wpack + ((cotile * 32 + lane) << 6);

#pragma unroll
    for (int t = 0; t < 9; ++t) {
        const int kh = t / 3, kw = t % 3;
        const unsigned char* bptr =
            actp + (((b * HP + h + kh) * WP + (w + kw)) << 8) + hi * 16;
        if (t + 1 < 9) {
            const int kh2 = (t + 1) / 3, kw2 = (t + 1) % 3;
            __builtin_prefetch(actp + (((b * HP + h + kh2) * WP + (w + kw2)) << 8), 0, 1);
        }
#pragma unroll
        for (int q = 0; q < 2; ++q) {
            v16i A = *(const v16i*)(wbase + ((t * 2 + q) << 14));
            const int4* bp = (const int4*)(bptr + (q << 7));
            int4 b0 = bp[0], b1 = bp[2], b2 = bp[4], b3 = bp[6];
            v16i Bf;
            Bf[0] = b0.x;  Bf[1] = b0.y;  Bf[2] = b0.z;  Bf[3] = b0.w;
            Bf[4] = b1.x;  Bf[5] = b1.y;  Bf[6] = b1.z;  Bf[7] = b1.w;
            Bf[8] = b2.x;  Bf[9] = b2.y;  Bf[10] = b2.z; Bf[11] = b2.w;
            Bf[12] = b3.x; Bf[13] = b3.y; Bf[14] = b3.z; Bf[15] = b3.w;
            c = __builtin_amdgcn_wmma_f32_16x16x128_fp8_fp8(A, Bf, (short)0, c,
                                                            false, false);
        }
    }

    const int px = (h << 6) + w;
#pragma unroll
    for (int g = 0; g < 8; ++g) {
        const int co = (cotile << 4) + g + (hi << 3);
        float v = c[g] * alpha[co] + beta[co];
        const float x0 = x[(((b << 8) + 2 * co) << 12) + px];
        const float x1 = x[(((b << 8) + 2 * co + 1) << 12) + px];
        v += 0.5f * (x0 + x1) + bias2[(b << 7) + co];
        v = v > 0.f ? v : prelu_a[co] * v;
        v += bias3[(b << 7) + co];
        out[(((b << 7) + co) << 12) + px] = v;
    }
}

extern "C" void kernel_launch(void* const* d_in, const int* in_sizes, int n_in,
                              void* d_out, int out_size, void* d_ws, size_t ws_size,
                              hipStream_t stream) {
    (void)in_sizes; (void)n_in; (void)out_size; (void)ws_size;
    const float* x     = (const float*)d_in[0];
    const float* emb   = (const float*)d_in[1];
    const float* m1w   = (const float*)d_in[2];
    const float* m1b   = (const float*)d_in[3];
    const float* convw = (const float*)d_in[4];
    const float* convb = (const float*)d_in[5];
    const float* gam   = (const float*)d_in[6];
    const float* bet   = (const float*)d_in[7];
    const float* mean  = (const float*)d_in[8];
    const float* var   = (const float*)d_in[9];
    const float* m2w   = (const float*)d_in[10];
    const float* m2b   = (const float*)d_in[11];
    const float* pre   = (const float*)d_in[12];
    const float* m3w   = (const float*)d_in[13];
    const float* m3b   = (const float*)d_in[14];
    float* out = (float*)d_out;

    char* ws = (char*)d_ws;
    float* silu  = (float*)(ws + OFF_SILU);
    float* bias1 = (float*)(ws + OFF_BIAS1);
    float* bias2 = (float*)(ws + OFF_BIAS2);
    float* bias3 = (float*)(ws + OFF_BIAS3);
    float* alpha = (float*)(ws + OFF_ALPHA);
    float* beta  = (float*)(ws + OFF_BETA);
    unsigned char* wpack = (unsigned char*)(ws + OFF_WPACK);
    unsigned char* actp  = (unsigned char*)(ws + OFF_ACTP);

    k_silu<<<32, 256, 0, stream>>>(emb, silu);
    k_bias<<<32, 256, 0, stream>>>(silu, m1w, m1b, m2w, m2b, m3w, m3b,
                                   bias1, bias2, bias3);
    k_scale<<<128, 256, 0, stream>>>(convw, convb, gam, bet, mean, var, alpha, beta);
    k_wpack<<<1152, 256, 0, stream>>>(convw, wpack);
    k_fill<<<(int)(ACTP_BYTES / 16 + 255) / 256, 256, 0, stream>>>((int4*)actp,
                                                                   (int)(ACTP_BYTES / 16));
    k_binarize<<<65536, 256, 0, stream>>>(x, bias1, actp);
    k_gemm<<<4096, 256, 0, stream>>>(actp, wpack, alpha, beta, bias2, bias3, pre, x, out);
}
